// Attention_56899726738131
// MI455X (gfx1250) — matrix operand order repaired and checked by reading, not compiled, not run on hardware
//
#include <hip/hip_runtime.h>
#include <hip/hip_fp16.h>
#include <math.h>

// CDNA5 / gfx1250 wave32 WMMA attention:
//   out = softmax(mask(QK^T/sqrt(d))) V, also emits full p_attn.
// B=4 H=16 L=1024 D=64, fp32 I/O, f16 WMMA with f32 accumulation.

typedef _Float16 v16h __attribute__((ext_vector_type(16)));
typedef float    v8f  __attribute__((ext_vector_type(8)));

static constexpr int   Bc = 4, Hc = 16, Lc = 1024, Dc = 64;
static constexpr float SCALE   = 0.125f;   // 1/sqrt(64)
static constexpr float NEG_INF = -1e9f;

// LDS partition (dynamic shared memory, ~199 KB < 320 KB per WGP)
static constexpr int    SC_STRIDE = Lc + 16;                 // fp32 scores row stride (bank-conflict pad)
static constexpr int    VT_STRIDE = Lc + 8;                  // f16 V^T row stride
static constexpr size_t SC_BYTES  = (size_t)16 * SC_STRIDE * 4;
static constexpr size_t VT_BYTES  = (size_t)Dc * VT_STRIDE * 2;
static constexpr size_t RED_BYTES = 16 * 16 * 4;
static constexpr size_t OAC_BYTES = 16 * Dc * 4;
static constexpr size_t SMEM_BYTES = SC_BYTES + VT_BYTES + RED_BYTES + OAC_BYTES;

__global__ __launch_bounds__(256, 1)
void Attention_56899726738131_kernel(const float* __restrict__ Q,
                                     const float* __restrict__ K,
                                     const float* __restrict__ V,
                                     const int*   __restrict__ M,
                                     float* __restrict__ Out,
                                     float* __restrict__ P)
{
    extern __shared__ unsigned char smem[];
    float*     sc   = (float*)smem;                                   // [16][SC_STRIDE] scores / probs
    _Float16*  vt   = (_Float16*)(smem + SC_BYTES);                   // [64][VT_STRIDE] V transposed f16
    float*     red  = (float*)(smem + SC_BYTES + VT_BYTES);           // [16][16] row reductions
    float*     oacc = (float*)(smem + SC_BYTES + VT_BYTES + RED_BYTES); // [16][64] split-K partial

    const int tid   = threadIdx.x;
    const int wave  = tid >> 5;        // 0..7 (wave32)
    const int lane  = tid & 31;
    const int nlo   = lane & 15;       // N index / M row within fragments
    const int khalf = lane >> 4;       // which K-half of the fragment this lane holds

    const int bid = blockIdx.x;
    const int qt  = bid & 63;          // query tile within (b,h)
    const int bh  = bid >> 6;          // b*H + h
    const int b   = bh >> 4;           // H == 16
    const int q0  = qt * 16;

    const float* Qb = Q + ((size_t)bh * Lc + q0) * Dc;
    const float* Kb = K + (size_t)bh * Lc * Dc;
    const float* Vb = V + (size_t)bh * Lc * Dc;
    const int*   Mb = M + ((size_t)b * Lc + q0) * Lc;   // mask[b,0,q0+m,col]
    float*       Pb = P + ((size_t)bh * Lc + q0) * Lc;
    float*       Ob = Out + ((size_t)bh * Lc + q0) * Dc;

    // ---------------- Stage V -> LDS transposed f16: vt[d][key] -----------------
    // Coalesced: consecutive tids read consecutive addresses of V[key][d].
    {
        const int d  = tid & 63;
        const int k0 = tid >> 6;       // 0..3
        #pragma unroll 4
        for (int it = 0; it < Lc; it += 4) {
            const int key = it + k0;
            vt[d * VT_STRIDE + key] = (_Float16)Vb[(size_t)key * Dc + d];
        }
    }

    // ---------------- Phase 1: S = (Q K^T) * scale, masked, into LDS ------------
    // A fragment layout (16-bit A 16x32): lane m=nlo; elems 0..7 -> K=khalf*8+i,
    // elems 8..15 -> K=16+khalf*8+i.
    v16h aq0, aq1;
    {
        const float* qr = Qb + (size_t)nlo * Dc + khalf * 8;
        #pragma unroll
        for (int i = 0; i < 8; ++i) {
            aq0[i]     = (_Float16)qr[i];
            aq0[8 + i] = (_Float16)qr[16 + i];
            aq1[i]     = (_Float16)qr[32 + i];
            aq1[8 + i] = (_Float16)qr[48 + i];
        }
    }

    // Each wave owns keys [wave*128, wave*128+128), 8 tiles of 16 keys.
    #pragma unroll
    for (int t = 0; t < 8; ++t) {
        const int kb = wave * 128 + t * 16;
        // B fragment (16-bit B 32x16): lane n=nlo -> key kb+nlo;
        // elems 0..7 -> K(d) = khalf*8 + e, elems 8..15 -> K(d) = 16 + khalf*8 + (e-8).
        const float* kr = Kb + (size_t)(kb + nlo) * Dc + khalf * 8;
        v16h bk0, bk1;
        #pragma unroll
        for (int e = 0; e < 16; ++e) {
            const int eo = e < 8 ? e : e + 8;
            bk0[e] = (_Float16)kr[eo];
            bk1[e] = (_Float16)kr[32 + eo];
        }

        v8f acc = {};
        acc = __builtin_amdgcn_wmma_f32_16x16x32_f16(false, aq0, false, bk0,
                                                     (short)0, acc, false, false);
        acc = __builtin_amdgcn_wmma_f32_16x16x32_f16(false, aq1, false, bk1,
                                                     (short)0, acc, false, false);

        // C/D layout: VGPR r, lanes0-15 -> M=r, lanes16-31 -> M=8+r; N=nlo.
        const int col = kb + nlo;
        #pragma unroll
        for (int r = 0; r < 8; ++r) {
            const int m = r + khalf * 8;
            const int msk = Mb[(size_t)m * Lc + col];
            sc[m * SC_STRIDE + col] = msk ? acc[r] * SCALE : NEG_INF;
        }
    }
    __syncthreads();

    // ---------------- Phase 2: row softmax + write p_attn -----------------------
    {
        const int r = tid >> 4;       // row 0..15
        const int c = tid & 15;       // 16 threads per row, interleaved cols
        float* row = sc + r * SC_STRIDE;

        float mx = -3.4e38f;
        #pragma unroll 8
        for (int j = 0; j < 64; ++j) mx = fmaxf(mx, row[c + 16 * j]);
        red[r * 16 + c] = mx;
        __syncthreads();
        mx = red[r * 16];
        #pragma unroll
        for (int i = 1; i < 16; ++i) mx = fmaxf(mx, red[r * 16 + i]);
        __syncthreads();

        float sum = 0.f;
        #pragma unroll 8
        for (int j = 0; j < 64; ++j) {
            const int cj = c + 16 * j;
            const float e = __expf(row[cj] - mx);  // masked (-1e9) underflows to exactly 0
            row[cj] = e;
            sum += e;
        }
        red[r * 16 + c] = sum;
        __syncthreads();
        sum = 0.f;
        #pragma unroll
        for (int i = 0; i < 16; ++i) sum += red[r * 16 + i];
        const float inv = 1.0f / sum;

        float* prow = Pb + (size_t)r * Lc;
        #pragma unroll 8
        for (int j = 0; j < 64; ++j) {
            const int cj = c + 16 * j;
            const float p = row[cj] * inv;
            row[cj]  = p;      // keep in LDS for PV matmul
            prow[cj] = p;      // emit p_attn
        }
    }
    __syncthreads();

    // ---------------- Phase 3: out = P V  (split-K over wave pairs) -------------
    // wave -> (ntile = wave&3 selects 16 output dims, ksel = wave>>2 selects key half)
    const int ntile = wave & 3;
    const int ksel  = wave >> 2;
    v8f oa = {};
    #pragma unroll
    for (int kt = 0; kt < 16; ++kt) {
        const int kk = ksel * 512 + kt * 32;
        // A = P[16 x 32] from LDS (f32 -> f16)
        const float* prow = sc + nlo * SC_STRIDE + kk + khalf * 8;
        v16h ap;
        #pragma unroll
        for (int i = 0; i < 8; ++i) {
            ap[i]     = (_Float16)prow[i];
            ap[8 + i] = (_Float16)prow[16 + i];
        }
        // B = V^T[32 keys x 16 dims] from transposed LDS: keys khalf*8+e and 16+khalf*8+e per lane
        const _Float16* vrow = vt + (ntile * 16 + nlo) * VT_STRIDE + kk + khalf * 8;
        v16h bv;
        #pragma unroll
        for (int e = 0; e < 16; ++e) bv[e] = vrow[e < 8 ? e : e + 8];

        oa = __builtin_amdgcn_wmma_f32_16x16x32_f16(false, ap, false, bv,
                                                    (short)0, oa, false, false);
    }

    // Combine the two key-halves through LDS, then write out.
    if (ksel == 0) {
        #pragma unroll
        for (int r = 0; r < 8; ++r) {
            const int m = r + khalf * 8;
            oacc[m * 64 + ntile * 16 + nlo] = oa[r];
        }
    }
    __syncthreads();
    if (ksel == 1) {
        #pragma unroll
        for (int r = 0; r < 8; ++r) {
            const int m = r + khalf * 8;
            const int d = ntile * 16 + nlo;
            Ob[(size_t)m * Dc + d] = oacc[m * 64 + d] + oa[r];
        }
    }
}

extern "C" void kernel_launch(void* const* d_in, const int* in_sizes, int n_in,
                              void* d_out, int out_size, void* d_ws, size_t ws_size,
                              hipStream_t stream) {
    (void)in_sizes; (void)n_in; (void)out_size; (void)d_ws; (void)ws_size;
    const float* Q = (const float*)d_in[0];
    const float* K = (const float*)d_in[1];
    const float* V = (const float*)d_in[2];
    const int*   M = (const int*)d_in[3];
    float* Out = (float*)d_out;                                  // [B,H,L,D]
    float* P   = (float*)d_out + (size_t)Bc * Hc * Lc * Dc;      // [B,H,L,L]

    // Raise dynamic-LDS cap (gfx1250 WGP allows up to 320 KB). Deterministic,
    // not a stream op -> graph-capture safe.
    (void)hipFuncSetAttribute((const void*)Attention_56899726738131_kernel,
                              hipFuncAttributeMaxDynamicSharedMemorySize,
                              (int)SMEM_BYTES);

    const dim3 grid(Bc * Hc * (Lc / 16));   // 4096 workgroups
    Attention_56899726738131_kernel<<<grid, 256, SMEM_BYTES, stream>>>(Q, K, V, M, Out, P);
}
